// NeighborMLPConvLayer_55490977465089
// MI455X (gfx1250) — matrix operand order repaired and checked by reading, not compiled, not run on hardware
//
#include <hip/hip_runtime.h>
#include <hip/hip_bf16.h>
#include <math.h>

// NeighborMLPConv: out[i] = mean_{j in nbr(i)} MLP(concat(x[j], x[i]))
// MLP = Linear(128->128) -> exact GELU -> Linear(128->64)
// bf16 WMMA (v_wmma_f32_16x16x32_bf16), 16-edge tiles per wave.

#define C_IN 64
#define HID 128
#define C_OUT 64

typedef __attribute__((ext_vector_type(16))) __bf16 v16bf;
typedef __attribute__((ext_vector_type(8)))  float  v8f;

__device__ __forceinline__ unsigned short f2bf(float f) {
  unsigned u = __float_as_uint(f);
  u += 0x7FFFu + ((u >> 16) & 1u);          // round-to-nearest-even
  return (unsigned short)(u >> 16);
}

// Branch-free exact-GELU: Abramowitz-Stegun 7.1.26 erf (|err| < 1.5e-7).
// Uses v_rcp_f32 / v_exp_f32 (TRANS pipe, co-executes with WMMA), no EXEC
// divergence (the ocml erff has a saveexec two-path region).
__device__ __forceinline__ float gelu_exact(float x) {
  float y = fabsf(x) * 0.70710678118654752f;       // |x|/sqrt(2)
  float t = __builtin_amdgcn_rcpf(fmaf(0.3275911f, y, 1.0f));
  float p = fmaf(1.061405429f, t, -1.453152027f);
  p = fmaf(p, t, 1.421413741f);
  p = fmaf(p, t, -0.284496736f);
  p = fmaf(p, t, 0.254829592f);
  p = p * t;
  float e = __expf(-y * y);
  float erf_abs = fmaf(-p, e, 1.0f);               // 1 - P(t)*exp(-y^2)
  float erf_x = copysignf(erf_abs, x);
  return 0.5f * x * (1.0f + erf_x);
}

union BfFrag {
  v16bf v;
  unsigned short us[16];
  uint4 q[2];
};

// ---------------------------------------------------------------------------
// Kernel 1: expand row_splits -> seg_ids, compute 1/max(count,1), zero output
// ---------------------------------------------------------------------------
__global__ void nmlp_init_kernel(const int* __restrict__ rs,
                                 int* __restrict__ seg,
                                 float* __restrict__ invc,
                                 float* __restrict__ out,
                                 int nN) {
  int i = blockIdx.x * blockDim.x + threadIdx.x;
  if (i >= nN) return;
  int s = rs[i], e = rs[i + 1];
  int cnt = e - s;
  invc[i] = 1.0f / (float)(cnt > 0 ? cnt : 1);
  for (int k = s; k < e; ++k) seg[k] = i;
  float4* o = (float4*)(out + (size_t)i * C_OUT);
#pragma unroll
  for (int k = 0; k < C_OUT / 4; ++k) o[k] = make_float4(0.f, 0.f, 0.f, 0.f);
}

// ---------------------------------------------------------------------------
// Kernel 2: per-edge MLP via WMMA, atomic mean-scatter into out
// ---------------------------------------------------------------------------
__launch_bounds__(128, 2)
__global__ void nmlp_conv_kernel(const float* __restrict__ feats,
                                 const float* __restrict__ W1,
                                 const float* __restrict__ b1,
                                 const float* __restrict__ W2,
                                 const float* __restrict__ b2,
                                 const int* __restrict__ nbr_idx,
                                 const int* __restrict__ seg_ids,
                                 const float* __restrict__ invc,
                                 float* __restrict__ out,
                                 int E, int ntiles) {
  // bf16 weights pre-swizzled to B-fragment layout:
  // w1s[((kb*8 + nt)*32 + lane)*8 + p] = pack(W1[k][n], W1[k+1][n])
  //   k = kb*32 + (lane>=16 ? 8 : 0) + (p>=4 ? 16 : 0) + 2(p&3) ,  n = nt*16 + (lane&15)
  __shared__ __align__(16) unsigned w1s[8192];               // 32 KB
  __shared__ __align__(16) unsigned w2s[4096];               // 16 KB
  // per-wave hidden staging, XOR-swizzled 8-elem chunks (conflict-free b128)
  __shared__ __align__(16) unsigned short Hst[4][16 * HID];  // 16 KB
  // total = 64 KB

  const int tid = threadIdx.x;

  for (int idx = tid; idx < 8192; idx += 128) {
    int p = idx & 7, lane = (idx >> 3) & 31, nt = (idx >> 8) & 7, kb = idx >> 11;
    int k = kb * 32 + ((lane & 16) ? 8 : 0) + ((p & 4) ? 16 : 0) + 2 * (p & 3);
    int n = nt * 16 + (lane & 15);
    unsigned lo = f2bf(W1[(size_t)k * HID + n]);
    unsigned hi = f2bf(W1[(size_t)(k + 1) * HID + n]);
    w1s[idx] = lo | (hi << 16);
  }
  for (int idx = tid; idx < 4096; idx += 128) {
    int p = idx & 7, lane = (idx >> 3) & 31, nt = (idx >> 8) & 3, kb = idx >> 10;
    int k = kb * 32 + ((lane & 16) ? 8 : 0) + ((p & 4) ? 16 : 0) + 2 * (p & 3);
    int n = nt * 16 + (lane & 15);
    unsigned lo = f2bf(W2[(size_t)k * C_OUT + n]);
    unsigned hi = f2bf(W2[(size_t)(k + 1) * C_OUT + n]);
    w2s[idx] = lo | (hi << 16);
  }
  __syncthreads();

  const int wv = tid >> 5;
  const int lane = tid & 31;
  const int r = lane & 15;               // A/C fragment row / col index
  const int half = (lane & 16) ? 8 : 0;  // K sub-offset for upper lane half

  unsigned short* Hw = Hst[wv];
  const char* fb = (const char*)feats;
  const int stride = gridDim.x * 4;

  for (int tile = blockIdx.x * 4 + wv; tile < ntiles; tile += stride) {
    int e = tile * 16 + r;
    if (e >= E) e = E - 1;               // E is a multiple of 16; safety clamp
    const int seg = seg_ids[e];
    const int nbr = nbr_idx[e];
    const float inv = invc[seg];

    // prefetch next tile's index words while we compute
    {
      int tn = tile + stride;
      if (tn < ntiles) {
        __builtin_prefetch(&seg_ids[tn * 16 + r], 0, 3);
        __builtin_prefetch(&nbr_idx[tn * 16 + r], 0, 3);
      }
    }

    // hoisted per-row seg/inv for the C-fragment rows this lane owns
    unsigned segOff[8];
    float invM[8];
#pragma unroll
    for (int v = 0; v < 8; ++v) {
      int M = v + half;
      segOff[v] = (unsigned)__shfl(seg, M, 32) * (C_OUT * 4u);
      invM[v] = __shfl(inv, M, 32);
    }

    // --- gather A fragments (16 edges x 128 bf16), saddr + 32-bit voffset --
    const unsigned offN = (unsigned)nbr * (C_IN * 4u) + (unsigned)(half * 4);
    const unsigned offS = (unsigned)seg * (C_IN * 4u) + (unsigned)(half * 4);
    const unsigned abase[4] = {offN, offN + 128u, offS, offS + 128u};
    BfFrag a[4];
#pragma unroll
    for (int kb = 0; kb < 4; ++kb) {
      float4 f0 = *(const float4*)(fb + abase[kb] + 0);    // K c0..c0+3
      float4 f1 = *(const float4*)(fb + abase[kb] + 16);   // K c0+4..c0+7
      float4 f2 = *(const float4*)(fb + abase[kb] + 64);   // K c1..c1+3
      float4 f3 = *(const float4*)(fb + abase[kb] + 80);   // K c1+4..c1+7
      a[kb].us[0] = f2bf(f0.x);  a[kb].us[1] = f2bf(f0.y);
      a[kb].us[2] = f2bf(f0.z);  a[kb].us[3] = f2bf(f0.w);
      a[kb].us[4] = f2bf(f1.x);  a[kb].us[5] = f2bf(f1.y);
      a[kb].us[6] = f2bf(f1.z);  a[kb].us[7] = f2bf(f1.w);
      a[kb].us[8] = f2bf(f2.x);  a[kb].us[9] = f2bf(f2.y);
      a[kb].us[10] = f2bf(f2.z); a[kb].us[11] = f2bf(f2.w);
      a[kb].us[12] = f2bf(f3.x); a[kb].us[13] = f2bf(f3.y);
      a[kb].us[14] = f2bf(f3.z); a[kb].us[15] = f2bf(f3.w);
    }

    // --- GEMM1: [16x128] x [128x128], zero-C WMMA chain, bias in epilogue --
#pragma unroll 2
    for (int nt = 0; nt < 8; ++nt) {
      float bias = b1[nt * 16 + r];      // issued early, consumed post-WMMA
      BfFrag bf[4];
#pragma unroll
      for (int kb = 0; kb < 4; ++kb) {   // all B-frags first: overlapped waits
        const uint4* bp = (const uint4*)&w1s[(((kb * 8 + nt) * 32) + lane) * 8];
        bf[kb].q[0] = bp[0];
        bf[kb].q[1] = bp[1];
      }
      v8f acc = {};                       // folds to inline SRC2=0
#pragma unroll
      for (int kb = 0; kb < 4; ++kb) {
        acc = __builtin_amdgcn_wmma_f32_16x16x32_bf16(
            false, a[kb].v, false, bf[kb].v, (short)0, acc, false, false);
      }
#pragma unroll
      for (int v = 0; v < 8; ++v) {
        float g = gelu_exact(acc[v] + bias);
        int row = v + half;                     // C-fragment M
        int col = nt * 16 + r;                  // C-fragment N
        int ccs = (((col >> 3) ^ row) << 3);    // swizzled 8-elem chunk
        Hw[row * HID + ccs + (col & 7)] = f2bf(g);
      }
    }

    // --- GEMM2: [16x128] x [128x64], scaled atomic scatter -----------------
#pragma unroll 2
    for (int nt = 0; nt < 4; ++nt) {
      float bias = b2[nt * 16 + r];
      BfFrag af[4], bf[4];
#pragma unroll
      for (int kb = 0; kb < 4; ++kb) {
        int c0 = kb * 32 + half;
        int cs0 = (((c0 >> 3) ^ r) << 3);
        int cs1 = ((((c0 + 16) >> 3) ^ r) << 3);
        af[kb].q[0] = *(const uint4*)&Hw[r * HID + cs0];
        af[kb].q[1] = *(const uint4*)&Hw[r * HID + cs1];
        const uint4* bp = (const uint4*)&w2s[(((kb * 4 + nt) * 32) + lane) * 8];
        bf[kb].q[0] = bp[0];
        bf[kb].q[1] = bp[1];
      }
      v8f acc = {};
#pragma unroll
      for (int kb = 0; kb < 4; ++kb) {
        acc = __builtin_amdgcn_wmma_f32_16x16x32_bf16(
            false, af[kb].v, false, bf[kb].v, (short)0, acc, false, false);
      }
      unsigned colOff = (unsigned)((nt * 16 + r) * 4);
#pragma unroll
      for (int v = 0; v < 8; ++v) {
        float* dst = (float*)((char*)out + segOff[v] + colOff);
        atomicAdd(dst, (acc[v] + bias) * invM[v]);
      }
    }
  }
}

// ---------------------------------------------------------------------------
extern "C" void kernel_launch(void* const* d_in, const int* in_sizes, int n_in,
                              void* d_out, int out_size, void* d_ws, size_t ws_size,
                              hipStream_t stream) {
  const float* feats = (const float*)d_in[0];
  const float* W1    = (const float*)d_in[1];
  const float* b1    = (const float*)d_in[2];
  const float* W2    = (const float*)d_in[3];
  const float* b2    = (const float*)d_in[4];
  const int*   nbr   = (const int*)d_in[5];
  const int*   rs    = (const int*)d_in[6];

  const int E  = in_sizes[5];
  const int Nn = in_sizes[6] - 1;
  float* out = (float*)d_out;

  int*   seg  = (int*)d_ws;
  size_t segBytes = (((size_t)E * sizeof(int)) + 255) & ~(size_t)255;
  float* invc = (float*)((char*)d_ws + segBytes);

  nmlp_init_kernel<<<(Nn + 255) / 256, 256, 0, stream>>>(rs, seg, invc, out, Nn);

  const int ntiles = (E + 15) / 16;
  const int blocks = 2048;
  nmlp_conv_kernel<<<blocks, 128, 0, stream>>>(feats, W1, b1, W2, b2,
                                               nbr, seg, invc, out, E, ntiles);
}
